// GraphAttentionLayer_24309514895501
// MI455X (gfx1250) — matrix operand order repaired and checked by reading, not compiled, not run on hardware
//
#include <hip/hip_runtime.h>
#include <math.h>

// ---------------- problem constants (from reference) ----------------
#define N_NODES 50000
#define N_EDGES 800000
#define E_TOTAL (N_EDGES + N_NODES)   // edges + self loops = 850000
#define IN_F    256
#define OUT_F   128
static constexpr float kAlpha = 0.2f;  // LeakyReLU slope

// ---------------- CDNA5 WMMA vector types ----------------
typedef __attribute__((ext_vector_type(16))) _Float16     v16h;
typedef __attribute__((ext_vector_type(8)))  float        v8f;
typedef __attribute__((ext_vector_type(8)))  unsigned int v8u;

// ============================================================
// K0: init output accumulator, per-node max (-inf) and denom
// ============================================================
__global__ void k_init(float* __restrict__ out, float* __restrict__ mmax,
                       float* __restrict__ denom) {
  int i = blockIdx.x * 256 + threadIdx.x;
  if (i < N_NODES * OUT_F) out[i] = 0.0f;
  if (i < N_NODES) { mmax[i] = -1e30f; denom[i] = 0.0f; }
}

// ============================================================
// K1: split h (fp32, [N][256]) into f16 hi/lo pair: x ~= hi + lo
// ============================================================
__global__ void k_cvt_h(const float* __restrict__ h,
                        _Float16* __restrict__ hi, _Float16* __restrict__ lo) {
  int i = blockIdx.x * 256 + threadIdx.x;
  if (i >= N_NODES * IN_F) return;
  float x = h[i];
  _Float16 xh = (_Float16)x;          // RNE
  hi[i] = xh;
  lo[i] = (_Float16)(x - (float)xh);  // residual
}

// K2: W [256][128] -> transposed hi/lo f16, layout [n][k] so that a dword at
// (n*128 + kp) holds W[2kp][n], W[2kp+1][n] (K-adjacent pair for fixed N).
__global__ void k_cvt_w(const float* __restrict__ W,
                        _Float16* __restrict__ hi, _Float16* __restrict__ lo) {
  int i = blockIdx.x * 256 + threadIdx.x;   // i = n*IN_F + k
  if (i >= OUT_F * IN_F) return;
  int n = i / IN_F, k = i % IN_F;
  float x = W[k * OUT_F + n];
  _Float16 xh = (_Float16)x;
  hi[i] = xh;
  lo[i] = (_Float16)(x - (float)xh);
}

// ============================================================
// K3: Wh = h @ W via v_wmma_f32_16x16x32_f16, two-term f16 split:
//     A*B ~= Ahi*Bhi + Ahi*Blo + Alo*Bhi   (f32 accumulation)
// Block = 256 thr = 8 waves; wave w -> rows [blk*128 + w*16, +16), all 128 cols.
// ============================================================
__global__ void __launch_bounds__(256)
k_gemm(const unsigned int* __restrict__ Ahi, const unsigned int* __restrict__ Alo,
       const unsigned int* __restrict__ Bhi, const unsigned int* __restrict__ Blo,
       float* __restrict__ Wh) {
  const int wave  = threadIdx.x >> 5;
  const int lane  = threadIdx.x & 31;
  const int mBase = blockIdx.x * 128 + wave * 16;
  if (mBase >= N_NODES) return;           // wave-uniform: EXEC stays all-ones
  const int m    = mBase + (lane & 15);   // A-matrix row for this lane
  const int half = lane >> 4;             // K-half selector (ISA 7.12.2 layouts)
  const int n0   = lane & 15;             // B/C column within tile

  v8f acc[8];
  #pragma unroll
  for (int t = 0; t < 8; ++t)
    acc[t] = v8f{0.f, 0.f, 0.f, 0.f, 0.f, 0.f, 0.f, 0.f};

  const unsigned int* arow_hi = Ahi + m * (IN_F / 2);  // dword = packed f16 pair
  const unsigned int* arow_lo = Alo + m * (IN_F / 2);

  #pragma unroll
  for (int kt = 0; kt < IN_F / 32; ++kt) {
    if (kt + 1 < IN_F / 32) {                        // global_prefetch_b8 next K-tile
      __builtin_prefetch(arow_hi + (kt + 1) * 16, 0, 1);
      __builtin_prefetch(arow_lo + (kt + 1) * 16, 0, 1);
    }
    // A fragment 16x32: element pair p -> K = (p>>2)*16 + half*8 + (p&3)*2
    v8u au_hi, au_lo;
    #pragma unroll
    for (int p = 0; p < 8; ++p) {
      int kp = kt * 16 + ((p >> 2) << 3) + (half << 2) + (p & 3);
      au_hi[p] = arow_hi[kp];
      au_lo[p] = arow_lo[kp];
    }
    v16h a_hi = __builtin_bit_cast(v16h, au_hi);
    v16h a_lo = __builtin_bit_cast(v16h, au_lo);

    #pragma unroll
    for (int t = 0; t < 8; ++t) {
      // B fragment 32x16: lane n = t*16+n0; element pair p -> K = (p>>2)*16 + half*8 + (p&3)*2
      const unsigned int* bcol_hi = Bhi + (t * 16 + n0) * (IN_F / 2);
      const unsigned int* bcol_lo = Blo + (t * 16 + n0) * (IN_F / 2);
      v8u bu_hi, bu_lo;
      #pragma unroll
      for (int p = 0; p < 8; ++p) {
        int kp = kt * 16 + ((p >> 2) << 3) + (half << 2) + (p & 3);
        bu_hi[p] = bcol_hi[kp];
        bu_lo[p] = bcol_lo[kp];
      }
      v16h b_hi = __builtin_bit_cast(v16h, bu_hi);
      v16h b_lo = __builtin_bit_cast(v16h, bu_lo);
      acc[t] = __builtin_amdgcn_wmma_f32_16x16x32_f16(false, a_hi, false, b_hi,
                                                      (short)0, acc[t], false, false);
      acc[t] = __builtin_amdgcn_wmma_f32_16x16x32_f16(false, a_hi, false, b_lo,
                                                      (short)0, acc[t], false, false);
      acc[t] = __builtin_amdgcn_wmma_f32_16x16x32_f16(false, a_lo, false, b_hi,
                                                      (short)0, acc[t], false, false);
    }
  }

  // C/D 16x16 f32 layout: lane n = n0, VGPR r -> M = r + 8*half
  #pragma unroll
  for (int t = 0; t < 8; ++t) {
    int n = t * 16 + n0;
    #pragma unroll
    for (int r = 0; r < 8; ++r) {
      int mm = mBase + r + (half << 3);
      Wh[mm * OUT_F + n] = acc[t][r];
    }
  }
}

// ============================================================
// K4: per-node attention scores: s_src = Wh@a[:128], s_dst = Wh@a[128:]
// one wave32 per node, shuffle reduction
// ============================================================
__global__ void k_scores(const float* __restrict__ Wh, const float* __restrict__ a,
                         float* __restrict__ ssrc, float* __restrict__ sdst) {
  int wave = threadIdx.x >> 5, lane = threadIdx.x & 31;
  int node = blockIdx.x * 8 + wave;
  if (node >= N_NODES) return;
  float p1 = 0.f, p2 = 0.f;
  #pragma unroll
  for (int c = lane; c < OUT_F; c += 32) {
    float w = Wh[node * OUT_F + c];
    p1 += w * a[c];
    p2 += w * a[OUT_F + c];
  }
  #pragma unroll
  for (int off = 16; off > 0; off >>= 1) {
    p1 += __shfl_xor(p1, off);
    p2 += __shfl_xor(p2, off);
  }
  if (lane == 0) { ssrc[node] = p1; sdst[node] = p2; }
}

// float atomic max via sign-split int trick (all values finite; init = -1e30)
__device__ __forceinline__ void atomic_max_float(float* addr, float v) {
  if (v >= 0.0f) atomicMax((int*)addr, __float_as_int(v));
  else           atomicMin((unsigned int*)addr, __float_as_uint(v));
}

// ============================================================
// K5: e = leakyrelu(s_dst[col] + s_src[row]); segment max over col
// ============================================================
__global__ void k_edge_max(const long long* __restrict__ adj,
                           const float* __restrict__ ssrc, const float* __restrict__ sdst,
                           float* __restrict__ ebuf, float* __restrict__ mmax) {
  int e = blockIdx.x * 256 + threadIdx.x;
  if (e >= E_TOTAL) return;
  int r, c;
  if (e < N_EDGES) { r = (int)adj[e]; c = (int)adj[N_EDGES + e]; }
  else             { r = c = e - N_EDGES; }          // appended self loop
  float x = sdst[c] + ssrc[r];
  x = x > 0.0f ? x : kAlpha * x;
  ebuf[e] = x;
  atomic_max_float(&mmax[c], x);
}

// K6: ex = exp(e - max[col]); segment sum over col
__global__ void k_edge_exp(const long long* __restrict__ adj,
                           const float* __restrict__ mmax,
                           float* __restrict__ ebuf, float* __restrict__ denom) {
  int e = blockIdx.x * 256 + threadIdx.x;
  if (e >= E_TOTAL) return;
  int c = (e < N_EDGES) ? (int)adj[N_EDGES + e] : e - N_EDGES;
  float ex = expf(ebuf[e] - mmax[c]);
  ebuf[e] = ex;                 // recomputed from scratch every call -> deterministic
  atomicAdd(&denom[c], ex);
}

// ============================================================
// K7: h_prime[col] += att * Wh[row]; 128 threads per edge (coalesced Wh row)
// ============================================================
__global__ void __launch_bounds__(256)
k_edge_agg(const long long* __restrict__ adj, const float* __restrict__ Wh,
           const float* __restrict__ ebuf, const float* __restrict__ denom,
           float* __restrict__ out) {
  int e = blockIdx.x * 2 + (threadIdx.x >> 7);
  if (e >= E_TOTAL) return;
  int f = threadIdx.x & 127;
  int r, c;
  if (e < N_EDGES) { r = (int)adj[e]; c = (int)adj[N_EDGES + e]; }
  else             { r = c = e - N_EDGES; }
  float att = ebuf[e] / denom[c];
  atomicAdd(&out[c * OUT_F + f], att * Wh[r * OUT_F + f]);
}

// K8: out = elu(out), in place
__global__ void k_elu(float* __restrict__ out) {
  int i = blockIdx.x * 256 + threadIdx.x;
  if (i >= N_NODES * OUT_F) return;
  float v = out[i];
  out[i] = v > 0.0f ? v : (expf(v) - 1.0f);
}

// ============================================================
extern "C" void kernel_launch(void* const* d_in, const int* in_sizes, int n_in,
                              void* d_out, int out_size, void* d_ws, size_t ws_size,
                              hipStream_t stream) {
  const float*     h   = (const float*)d_in[0];
  const long long* adj = (const long long*)d_in[1];   // int64 [2][E]
  const float*     W   = (const float*)d_in[2];
  const float*     a   = (const float*)d_in[3];
  float*           out = (float*)d_out;               // [N][128] fp32

  // ---- carve workspace (~81 MB) ----
  char* p = (char*)d_ws;
  auto carve = [&](size_t bytes) -> char* {
    char* q = p;
    p += (bytes + 255) & ~(size_t)255;
    return q;
  };
  _Float16* h_hi  = (_Float16*)carve(sizeof(_Float16) * (size_t)N_NODES * IN_F);
  _Float16* h_lo  = (_Float16*)carve(sizeof(_Float16) * (size_t)N_NODES * IN_F);
  _Float16* wt_hi = (_Float16*)carve(sizeof(_Float16) * (size_t)OUT_F * IN_F);
  _Float16* wt_lo = (_Float16*)carve(sizeof(_Float16) * (size_t)OUT_F * IN_F);
  float*    Wh    = (float*)   carve(sizeof(float) * (size_t)N_NODES * OUT_F);
  float*    ssrc  = (float*)   carve(sizeof(float) * N_NODES);
  float*    sdst  = (float*)   carve(sizeof(float) * N_NODES);
  float*    mmax  = (float*)   carve(sizeof(float) * N_NODES);
  float*    denom = (float*)   carve(sizeof(float) * N_NODES);
  float*    ebuf  = (float*)   carve(sizeof(float) * E_TOTAL);

  // ---- pipeline ----
  k_init <<<(N_NODES * OUT_F + 255) / 256, 256, 0, stream>>>(out, mmax, denom);
  k_cvt_h<<<(N_NODES * IN_F + 255) / 256, 256, 0, stream>>>(h, h_hi, h_lo);
  k_cvt_w<<<(OUT_F * IN_F + 255) / 256, 256, 0, stream>>>(W, wt_hi, wt_lo);
  k_gemm <<<(N_NODES + 127) / 128, 256, 0, stream>>>(
      (const unsigned int*)h_hi, (const unsigned int*)h_lo,
      (const unsigned int*)wt_hi, (const unsigned int*)wt_lo, Wh);
  k_scores<<<(N_NODES + 7) / 8, 256, 0, stream>>>(Wh, a, ssrc, sdst);
  k_edge_max<<<(E_TOTAL + 255) / 256, 256, 0, stream>>>(adj, ssrc, sdst, ebuf, mmax);
  k_edge_exp<<<(E_TOTAL + 255) / 256, 256, 0, stream>>>(adj, mmax, ebuf, denom);
  k_edge_agg<<<(E_TOTAL + 1) / 2, 256, 0, stream>>>(adj, Wh, ebuf, denom, out);
  k_elu<<<(N_NODES * OUT_F + 255) / 256, 256, 0, stream>>>(out);
}
